// CausalSelfAttention_76605036691607
// MI455X (gfx1250) — compile-verified, matrix operand order checked by reading, not run on hardware
//
#include <hip/hip_runtime.h>
#include <hip/hip_bf16.h>

// ---------------------------------------------------------------------------
// Types for WMMA fragments (CDNA5 / gfx1250, wave32)
// ---------------------------------------------------------------------------
typedef __attribute__((ext_vector_type(16))) __bf16 v16bf;
typedef __attribute__((ext_vector_type(8)))  float  v8f;

union FragBF16 {
    v16bf v;
    uint4 q[2];   // two 16-byte chunks = 2 x 8 bf16
};

__device__ __forceinline__ unsigned short f2bf(float x) {
    union { float f; unsigned int u; } c;
    c.f = x;
    unsigned int r = c.u + 0x7FFFu + ((c.u >> 16) & 1u);  // round-to-nearest-even
    return (unsigned short)(r >> 16);
}

// ---------------------------------------------------------------------------
// Problem constants (match reference: B=4, L=2048, D=1024, H=16, d_k=64)
// ---------------------------------------------------------------------------
#define BATCH   4
#define SEQ     2048
#define DMODEL  1024
#define NHEADS  16
#define DK      64
#define MROWS   (BATCH * SEQ)      // 8192
#define PFD     256                // prefetch distance: 8 k-steps * 32 elems

// ---------------------------------------------------------------------------
// fp32 -> bf16 conversion
// ---------------------------------------------------------------------------
__global__ void cvt_bf16_kernel(const float* __restrict__ src,
                                unsigned short* __restrict__ dst, int n) {
    int i = blockIdx.x * blockDim.x + threadIdx.x;
    if (i < n) dst[i] = f2bf(src[i]);
}

// ---------------------------------------------------------------------------
// QKV projection GEMM: C[m,n] = sum_k Xbf[m,k] * Wbf[n,k]
// M=8192, N=3072, K=1024.
// Block = 8 waves (4 M x 2 N), wave = 32x64 tile (2x4 WMMA frags).
// Epilogue scatters bf16 into Q [B,H,L,64], K [B,H,L,64], Vt [B,H,64,L].
// ---------------------------------------------------------------------------
__global__ __launch_bounds__(256) void gemm_qkv_kernel(
    const unsigned short* __restrict__ A,   // x bf16 [8192,1024]
    const unsigned short* __restrict__ Bm,  // W_qkv bf16 [3072,1024]
    unsigned short* __restrict__ Qo,
    unsigned short* __restrict__ Ko,
    unsigned short* __restrict__ Vt) {
    const int K = DMODEL;
    const int lane = threadIdx.x & 31;
    const int wave = threadIdx.x >> 5;
    const int wm = wave & 3;          // 0..3
    const int wn = wave >> 2;         // 0..1
    const int m0 = blockIdx.x * 128 + wm * 32;
    const int n0 = blockIdx.y * 128 + wn * 64;
    const int hl = lane >> 4;         // half-wave id
    const int lr = lane & 15;         // lane-in-half
    const int kbase = hl * 8;

    v8f acc[2][4] = {};

    for (int k = 0; k < K; k += 32) {
        FragBF16 af[2], bf[4];
#pragma unroll
        for (int t = 0; t < 2; ++t) {
            const unsigned short* ap = A + (size_t)(m0 + t * 16 + lr) * K + k + kbase;
            af[t].q[0] = *(const uint4*)(ap);
            af[t].q[1] = *(const uint4*)(ap + 16);
            if (k + PFD < K) __builtin_prefetch(ap + PFD, 0, 3);  // global_prefetch_b8
        }
#pragma unroll
        for (int t = 0; t < 4; ++t) {
            const unsigned short* bp = Bm + (size_t)(n0 + t * 16 + lr) * K + k + kbase;
            bf[t].q[0] = *(const uint4*)(bp);
            bf[t].q[1] = *(const uint4*)(bp + 16);
            if (k + PFD < K) __builtin_prefetch(bp + PFD, 0, 3);
        }
#pragma unroll
        for (int i = 0; i < 2; ++i)
#pragma unroll
            for (int j = 0; j < 4; ++j)
                acc[i][j] = __builtin_amdgcn_wmma_f32_16x16x32_bf16(
                    false, af[i].v, false, bf[j].v, (short)0, acc[i][j], false, false);
    }

    // Scatter epilogue: n = s*1024 + h*64 + d   (s in {0,1,2})
#pragma unroll
    for (int i = 0; i < 2; ++i)
#pragma unroll
        for (int j = 0; j < 4; ++j)
#pragma unroll
            for (int r = 0; r < 8; ++r) {
                int gm = m0 + i * 16 + r + 8 * hl;
                int gn = n0 + j * 16 + lr;
                unsigned short v = f2bf(acc[i][j][r]);
                int bb  = gm >> 11;        // / 2048
                int ll  = gm & (SEQ - 1);
                int s   = gn >> 10;        // 0=Q 1=K 2=V
                int rem = gn & 1023;
                int hh  = rem >> 6;
                int dd  = rem & 63;
                size_t bh = (size_t)bb * NHEADS + hh;
                if (s == 0)
                    Qo[(bh * SEQ + ll) * DK + dd] = v;
                else if (s == 1)
                    Ko[(bh * SEQ + ll) * DK + dd] = v;
                else
                    Vt[(bh * DK + dd) * SEQ + ll] = v;
            }
}

// ---------------------------------------------------------------------------
// Flash attention (causal). One wave per workgroup; one 16-row Q tile/wave.
// Key blocks of 32 -> P tile is 16x32 bf16 == one WMMA K-step.
// ---------------------------------------------------------------------------
__global__ __launch_bounds__(32) void attn_kernel(
    const unsigned short* __restrict__ Qg,   // [B*H, L, 64]
    const unsigned short* __restrict__ Kg,   // [B*H, L, 64]
    const unsigned short* __restrict__ Vtg,  // [B*H, 64, L]
    unsigned short* __restrict__ Og) {       // [B, L, H*64] bf16
    const int qt = blockIdx.x;               // 0..127
    const int bh = blockIdx.y;               // 0..63
    const int b  = bh >> 4;
    const int h  = bh & 15;
    const int lane = threadIdx.x;
    const int hl = lane >> 4;
    const int lr = lane & 15;
    const int kbase = hl * 8;
    const int q0 = qt * 16;

    __shared__ __align__(16) unsigned short Plds[16 * 40];  // 16 rows, stride 40

    // Load Q fragments (rows q0..q0+15, k=0..31 and 32..63)
    FragBF16 qf[2];
    {
        const unsigned short* qp = Qg + ((size_t)bh * SEQ + q0 + lr) * DK;
#pragma unroll
        for (int t = 0; t < 2; ++t) {
            qf[t].q[0] = *(const uint4*)(qp + t * 32 + kbase);
            qf[t].q[1] = *(const uint4*)(qp + t * 32 + kbase + 16);
        }
    }

    float mrow[8], lsum[8];
    v8f acc[4] = {};
#pragma unroll
    for (int r = 0; r < 8; ++r) { mrow[r] = -__builtin_inff(); lsum[r] = 0.0f; }

    const float scale = 0.125f;  // 1/sqrt(64)
    const int kmax = q0 + 16;    // exclusive causal key bound for this tile

    for (int kb0 = 0; kb0 < kmax; kb0 += 32) {
        // ---- S = Q @ K^T for two 16-wide key sub-tiles ----
        v8f s[2] = {};
#pragma unroll
        for (int j = 0; j < 2; ++j) {
            const unsigned short* kp = Kg + ((size_t)bh * SEQ + kb0 + j * 16 + lr) * DK;
            FragBF16 kf0, kf1;
            kf0.q[0] = *(const uint4*)(kp + kbase);
            kf0.q[1] = *(const uint4*)(kp + kbase + 16);
            kf1.q[0] = *(const uint4*)(kp + 32 + kbase);
            kf1.q[1] = *(const uint4*)(kp + 32 + kbase + 16);
            if (kb0 + 32 < kmax) __builtin_prefetch(kp + 32 * DK, 0, 3);  // next key block
            s[j] = __builtin_amdgcn_wmma_f32_16x16x32_bf16(
                false, qf[0].v, false, kf0.v, (short)0, s[j], false, false);
            s[j] = __builtin_amdgcn_wmma_f32_16x16x32_bf16(
                false, qf[1].v, false, kf1.v, (short)0, s[j], false, false);
        }

        // ---- scale + causal mask + online softmax ----
        float sv0[8], sv1[8], tmax[8];
#pragma unroll
        for (int r = 0; r < 8; ++r) {
            int row = q0 + r + 8 * hl;
            int c0 = kb0 + lr;
            int c1 = kb0 + 16 + lr;
            float a = s[0][r] * scale;
            float bvl = s[1][r] * scale;
            sv0[r] = (c0 <= row) ? a   : -__builtin_inff();
            sv1[r] = (c1 <= row) ? bvl : -__builtin_inff();
            tmax[r] = fmaxf(sv0[r], sv1[r]);
        }
#pragma unroll
        for (int off = 1; off < 16; off <<= 1)
#pragma unroll
            for (int r = 0; r < 8; ++r)
                tmax[r] = fmaxf(tmax[r], __shfl_xor(tmax[r], off, 32));

        float psum[8];
#pragma unroll
        for (int r = 0; r < 8; ++r) {
            float mnew = fmaxf(mrow[r], tmax[r]);
            float corr = __expf(mrow[r] - mnew);   // exp(-inf) = 0 first pass
            lsum[r] *= corr;
#pragma unroll
            for (int jn = 0; jn < 4; ++jn) acc[jn][r] *= corr;
            mrow[r] = mnew;
            float p0 = __expf(sv0[r] - mnew);
            float p1 = __expf(sv1[r] - mnew);
            psum[r] = p0 + p1;
            int rloc = r + 8 * hl;
            Plds[rloc * 40 + lr]      = f2bf(p0);
            Plds[rloc * 40 + 16 + lr] = f2bf(p1);
        }
#pragma unroll
        for (int off = 1; off < 16; off <<= 1)
#pragma unroll
            for (int r = 0; r < 8; ++r)
                psum[r] += __shfl_xor(psum[r], off, 32);
#pragma unroll
        for (int r = 0; r < 8; ++r) lsum[r] += psum[r];

        __syncthreads();  // single-wave WG: LDS store->load fence

        // ---- P fragment (A-layout) from LDS ----
        FragBF16 pf;
        {
            const unsigned short* pp = &Plds[lr * 40 + kbase];
            pf.q[0] = *(const uint4*)(pp);
            pf.q[1] = *(const uint4*)(pp + 16);
        }
        __syncthreads();  // protect Plds before next iteration's stores

        // ---- O += P @ V  (V pre-transposed: rows contiguous along keys) ----
#pragma unroll
        for (int jn = 0; jn < 4; ++jn) {
            const unsigned short* vp =
                Vtg + ((size_t)bh * DK + jn * 16 + lr) * SEQ + kb0 + kbase;
            FragBF16 vf;
            vf.q[0] = *(const uint4*)(vp);
            vf.q[1] = *(const uint4*)(vp + 16);
            if (kb0 + 32 < kmax) __builtin_prefetch(vp + 32, 0, 3);  // next key block
            acc[jn] = __builtin_amdgcn_wmma_f32_16x16x32_bf16(
                false, pf.v, false, vf.v, (short)0, acc[jn], false, false);
        }
    }

    // ---- normalize + store O (bf16, [B, L, H*64]) ----
#pragma unroll
    for (int r = 0; r < 8; ++r) {
        float inv = 1.0f / lsum[r];
        int row = q0 + r + 8 * hl;
        size_t ob = ((size_t)b * SEQ + row) * DMODEL + h * DK;
#pragma unroll
        for (int jn = 0; jn < 4; ++jn)
            Og[ob + jn * 16 + lr] = f2bf(acc[jn][r] * inv);
    }
}

// ---------------------------------------------------------------------------
// Output projection GEMM: out[m,n] = sum_k O[m,k] * Wo[n,k]  (fp32 out)
// M=8192, N=1024, K=1024.  Same 32x64 wave tile.
// ---------------------------------------------------------------------------
__global__ __launch_bounds__(256) void gemm_out_kernel(
    const unsigned short* __restrict__ A,   // O bf16 [8192,1024]
    const unsigned short* __restrict__ Bm,  // W_o bf16 [1024,1024]
    float* __restrict__ C) {
    const int K = DMODEL;
    const int lane = threadIdx.x & 31;
    const int wave = threadIdx.x >> 5;
    const int wm = wave & 3;
    const int wn = wave >> 2;
    const int m0 = blockIdx.x * 128 + wm * 32;
    const int n0 = blockIdx.y * 128 + wn * 64;
    const int hl = lane >> 4;
    const int lr = lane & 15;
    const int kbase = hl * 8;

    v8f acc[2][4] = {};

    for (int k = 0; k < K; k += 32) {
        FragBF16 af[2], bf[4];
#pragma unroll
        for (int t = 0; t < 2; ++t) {
            const unsigned short* ap = A + (size_t)(m0 + t * 16 + lr) * K + k + kbase;
            af[t].q[0] = *(const uint4*)(ap);
            af[t].q[1] = *(const uint4*)(ap + 16);
            if (k + PFD < K) __builtin_prefetch(ap + PFD, 0, 3);
        }
#pragma unroll
        for (int t = 0; t < 4; ++t) {
            const unsigned short* bp = Bm + (size_t)(n0 + t * 16 + lr) * K + k + kbase;
            bf[t].q[0] = *(const uint4*)(bp);
            bf[t].q[1] = *(const uint4*)(bp + 16);
            if (k + PFD < K) __builtin_prefetch(bp + PFD, 0, 3);
        }
#pragma unroll
        for (int i = 0; i < 2; ++i)
#pragma unroll
            for (int j = 0; j < 4; ++j)
                acc[i][j] = __builtin_amdgcn_wmma_f32_16x16x32_bf16(
                    false, af[i].v, false, bf[j].v, (short)0, acc[i][j], false, false);
    }

#pragma unroll
    for (int i = 0; i < 2; ++i)
#pragma unroll
        for (int j = 0; j < 4; ++j)
#pragma unroll
            for (int r = 0; r < 8; ++r) {
                int gm = m0 + i * 16 + r + 8 * hl;
                int gn = n0 + j * 16 + lr;
                C[(size_t)gm * DMODEL + gn] = acc[i][j][r];
            }
}

// ---------------------------------------------------------------------------
// Host launcher
// ---------------------------------------------------------------------------
extern "C" void kernel_launch(void* const* d_in, const int* in_sizes, int n_in,
                              void* d_out, int out_size, void* d_ws, size_t ws_size,
                              hipStream_t stream) {
    const float* x     = (const float*)d_in[0];  // [4,2048,1024]
    const float* W_qkv = (const float*)d_in[1];  // [3072,1024]
    const float* W_o   = (const float*)d_in[2];  // [1024,1024]
    float* out = (float*)d_out;

    const size_t N_X    = (size_t)MROWS * DMODEL;        // 8388608
    const size_t N_WQKV = (size_t)3 * DMODEL * DMODEL;   // 3145728
    const size_t N_WO   = (size_t)DMODEL * DMODEL;       // 1048576
    const size_t N_QKV1 = (size_t)BATCH * NHEADS * SEQ * DK;  // 8388608 each

    char* ws = (char*)d_ws;
    size_t off = 0;
    unsigned short* xb   = (unsigned short*)(ws + off); off += N_X    * 2;
    unsigned short* wqb  = (unsigned short*)(ws + off); off += N_WQKV * 2;
    unsigned short* wob  = (unsigned short*)(ws + off); off += N_WO   * 2;
    unsigned short* Qw   = (unsigned short*)(ws + off); off += N_QKV1 * 2;
    unsigned short* Kw   = (unsigned short*)(ws + off); off += N_QKV1 * 2;
    unsigned short* Vtw  = (unsigned short*)(ws + off); off += N_QKV1 * 2;
    unsigned short* Ow   = (unsigned short*)(ws + off); off += N_X    * 2;
    (void)ws_size; (void)off; (void)in_sizes; (void)n_in; (void)out_size;

    // 1) fp32 -> bf16 conversions
    {
        int t = 256;
        cvt_bf16_kernel<<<(int)((N_X + t - 1) / t), t, 0, stream>>>(x, xb, (int)N_X);
        cvt_bf16_kernel<<<(int)((N_WQKV + t - 1) / t), t, 0, stream>>>(W_qkv, wqb, (int)N_WQKV);
        cvt_bf16_kernel<<<(int)((N_WO + t - 1) / t), t, 0, stream>>>(W_o, wob, (int)N_WO);
    }

    // 2) QKV projection (scatters Q, K, V^T)
    gemm_qkv_kernel<<<dim3(MROWS / 128, (3 * DMODEL) / 128), 256, 0, stream>>>(
        xb, wqb, Qw, Kw, Vtw);

    // 3) causal flash attention
    attn_kernel<<<dim3(SEQ / 16, BATCH * NHEADS), 32, 0, stream>>>(Qw, Kw, Vtw, Ow);

    // 4) output projection (fp32 result)
    gemm_out_kernel<<<dim3(MROWS / 128, DMODEL / 128), 256, 0, stream>>>(Ow, wob, out);
}
